// HungarianMatcher_64415919505539
// MI455X (gfx1250) — matrix operand order repaired and checked by reading, not compiled, not run on hardware
//
#include <hip/hip_runtime.h>
#include <hip/hip_bf16.h>
#include <stdint.h>

// ---------------------------------------------------------------------------
// Problem constants (from reference): outputs [4,100,256,256], targets [4,20,256,256]
// ---------------------------------------------------------------------------
#define BB   4
#define NN   100
#define MM   20
#define HW   65536
#define NT   7                 // ceil(100/16) n-tiles
#define KS   16                // K slices (independent partial accumulators)
#define KBLK (HW / KS)         // 4096 K per workgroup
#define WPB  8                 // waves per block (256 threads, wave32)
#define KW   (KBLK / WPB)      // 512 K per wave
#define STEPS (KW / 32)        // 16 wmma k-steps per wave

typedef __attribute__((ext_vector_type(16))) _Float16     v16h;
typedef __attribute__((ext_vector_type(8)))  float        v8f;
typedef __attribute__((ext_vector_type(4)))  float        v4f;

static __device__ inline v8f zero8() {
    v8f z;
#pragma unroll
    for (int i = 0; i < 8; ++i) z[i] = 0.0f;
    return z;
}

// fused sigmoid / softplus / focal terms from a single exp+rcp+log
static __device__ inline void focal3(float xv, _Float16& fp, _Float16& fn, _Float16& pr) {
    float e   = __expf(xv);
    float t   = 1.0f + e;
    float ri  = __builtin_amdgcn_rcpf(t);   // 1/(1+e) == 1 - prob
    float p   = e * ri;                     // sigmoid
    float sp  = __logf(t);                  // softplus(x)
    float fpos = 0.25f * ri * ri * (sp - xv);   // ALPHA*(1-p)^2*softplus(-x)
    float fneg = 0.75f * p * p * sp;            // (1-ALPHA)*p^2*softplus(x)
    fp = (_Float16)fpos;
    fn = (_Float16)fneg;
    pr = (_Float16)p;
}

// ---------------------------------------------------------------------------
// Kernel 1: per-(b,m) target row sums (dice denominator term), no atomics
// ---------------------------------------------------------------------------
__global__ __launch_bounds__(256) void cost_tgtsum(const float* __restrict__ tgt,
                                                   float* __restrict__ stgt) {
    __shared__ float sh[256];
    const int bm = blockIdx.x;                       // 0 .. BB*MM-1
    const float* p = tgt + (size_t)bm * HW;
    float s = 0.0f;
    for (int i = threadIdx.x; i < HW; i += 256) s += p[i];
    sh[threadIdx.x] = s;
    __syncthreads();
    for (int o = 128; o > 0; o >>= 1) {
        if (threadIdx.x < o) sh[threadIdx.x] += sh[threadIdx.x + o];
        __syncthreads();
    }
    if (threadIdx.x == 0) stgt[bm] = sh[0];
}

// ---------------------------------------------------------------------------
// Kernel 2: main fused focal+dice contraction via v_wmma_f32_16x16x32_f16.
// grid = BB * NT * KS blocks of 256 threads (8 waves), each wave independent
// (no LDS / no barriers in the k-loop: A operands are loaded from global
//  directly in the WMMA A-layout, transcendentals fused in registers).
// Per wave accumulators (16n x 32m, f32):
//   P = focal_pos . T      N = focal_neg . (1-T)      Q = prob . T
// B-matrix column 20 is all-ones -> Q[:,20] = sum(prob) for free.
// ---------------------------------------------------------------------------
__global__ __launch_bounds__(256, 1) void cost_main(const float* __restrict__ x,
                                                    const float* __restrict__ tgt,
                                                    float* __restrict__ partial) {
    __shared__ float smem[WPB * 1536];               // 48 KB, end-of-kernel reduction only

    const int tid  = threadIdx.x;
    const int w    = tid >> 5;
    const int lane = tid & 31;

    const int bx = blockIdx.x;
    const int ks = bx % KS;
    const int nt = (bx / KS) % NT;
    const int b  = bx / (KS * NT);
    const int n0 = nt * 16;

    // ---- A loader: 16-bit A 16x32 WMMA layout, loaded straight from global ----
    // lane l: row = l&15; elems 0..15 = K koff..koff+15 (contiguous)
    const int arow = lane & 15;
    const int koff = (lane >> 4) << 4;                        // 0 or 16
    int nrow = n0 + arow; if (nrow >= NN) nrow = NN - 1;      // clamp; discarded later
    const float* xrow = x + ((size_t)b * NN + nrow) * (size_t)HW;

    // ---- B loader: lane = column, K contiguous: lanes0-15 K=0..15, lanes16-31 K=16..31
    const int mc = lane & 15;
    const int kh = (lane >> 4) << 4;
    const float* t0row = tgt + ((size_t)b * MM + mc) * (size_t)HW;          // m = mc (<16, valid)
    const int   m1     = 16 + mc;
    const float* t1row = tgt + ((size_t)b * MM + (m1 < MM ? m1 : 0)) * (size_t)HW;
    const bool  m1valid = (m1 < MM);
    const bool  m1one   = (m1 == MM);                                        // ones column

    v8f accP0 = zero8(), accP1 = zero8();
    v8f accN0 = zero8(), accN1 = zero8();
    v8f accQ0 = zero8(), accQ1 = zero8();

    int k = ks * KBLK + w * KW;
#pragma unroll 1
    for (int s = 0; s < STEPS; ++s, k += 32) {
        // ---- A: load 16 logits in WMMA layout, fuse transcendentals in regs ----
        const float* ap = xrow + k + koff;
        v4f a0 = *(const v4f*)(ap);
        v4f a1 = *(const v4f*)(ap + 4);
        v4f a2 = *(const v4f*)(ap + 8);
        v4f a3 = *(const v4f*)(ap + 12);
        float xv[16];
#pragma unroll
        for (int i = 0; i < 4; ++i) { xv[i] = a0[i]; xv[4+i] = a1[i]; xv[8+i] = a2[i]; xv[12+i] = a3[i]; }

        v16h Afp, Afn, Apr;
#pragma unroll
        for (int i = 0; i < 16; ++i) {
            _Float16 fp, fn, pr;
            focal3(xv[i], fp, fn, pr);
            Afp[i] = fp; Afn[i] = fn; Apr[i] = pr;
        }

        // ---- B tiles straight from global (K-contiguous per lane) ----
        v16h Bt0, Bt1;
        {
            const float* p0 = t0row + k + kh;
#pragma unroll
            for (int i = 0; i < 4; ++i) {
                v4f f = *(const v4f*)(p0 + 4 * i);
                for (int j = 0; j < 4; ++j) Bt0[4*i + j] = (_Float16)f[j];
            }
            if (m1valid) {
                const float* p1 = t1row + k + kh;
#pragma unroll
                for (int i = 0; i < 4; ++i) {
                    v4f f = *(const v4f*)(p1 + 4 * i);
                    for (int j = 0; j < 4; ++j) Bt1[4*i + j] = (_Float16)f[j];
                }
            } else {
                _Float16 c = m1one ? (_Float16)1.0f : (_Float16)0.0f;
#pragma unroll
                for (int i = 0; i < 16; ++i) Bt1[i] = c;
            }
        }
        v16h Bn0, Bn1;
#pragma unroll
        for (int i = 0; i < 16; ++i) { Bn0[i] = (_Float16)1.0f - Bt0[i]; Bn1[i] = (_Float16)1.0f - Bt1[i]; }

        // ---- 6 WMMAs: D = A x B + C, f32 accumulate ----
        accP0 = __builtin_amdgcn_wmma_f32_16x16x32_f16(false, Afp, false, Bt0, (short)0, accP0, false, false);
        accP1 = __builtin_amdgcn_wmma_f32_16x16x32_f16(false, Afp, false, Bt1, (short)0, accP1, false, false);
        accN0 = __builtin_amdgcn_wmma_f32_16x16x32_f16(false, Afn, false, Bn0, (short)0, accN0, false, false);
        accN1 = __builtin_amdgcn_wmma_f32_16x16x32_f16(false, Afn, false, Bn1, (short)0, accN1, false, false);
        accQ0 = __builtin_amdgcn_wmma_f32_16x16x32_f16(false, Apr, false, Bt0, (short)0, accQ0, false, false);
        accQ1 = __builtin_amdgcn_wmma_f32_16x16x32_f16(false, Apr, false, Bt1, (short)0, accQ1, false, false);
    }

    // ---- cross-wave reduction via LDS (only sync point), store partials ----
    {
        const int base = w * 1536 + lane * 8;
#pragma unroll
        for (int j = 0; j < 8; ++j) {
            smem[base + 0 * 256 + j] = accP0[j];
            smem[base + 1 * 256 + j] = accP1[j];
            smem[base + 2 * 256 + j] = accN0[j];
            smem[base + 3 * 256 + j] = accN1[j];
            smem[base + 4 * 256 + j] = accQ0[j];
            smem[base + 5 * 256 + j] = accQ1[j];
        }
    }
    __syncthreads();

    float* pout = partial + (size_t)((ks * BB + b) * NT + nt) * 1536;
    for (int e = tid; e < 1536; e += 256) {
        float s = 0.0f;
#pragma unroll
        for (int ww = 0; ww < WPB; ++ww) s += smem[ww * 1536 + e];
        pout[e] = s;
    }
}

// ---------------------------------------------------------------------------
// Kernel 3: fold KS partials into final cost C[b,n,m]
// C/D element (row nr, col mcol) lives at lane = mcol + 16*(nr>>3), vgpr j = nr&7
// ---------------------------------------------------------------------------
__global__ __launch_bounds__(256) void cost_finalize(const float* __restrict__ partial,
                                                     const float* __restrict__ stgt,
                                                     float* __restrict__ out) {
    const int t = blockIdx.x * 256 + threadIdx.x;
    if (t >= BB * NN * MM) return;
    const int m = t % MM;
    const int n = (t / MM) % NN;
    const int b = t / (MM * NN);
    const int nt = n >> 4;
    const int nr = n & 15;

    const int tile  = (m >= 16) ? 1 : 0;
    const int mcol  = m & 15;
    const int elem  = (mcol + ((nr >> 3) << 4)) * 8 + (nr & 7);
    const int elemS = (4 + ((nr >> 3) << 4)) * 8 + (nr & 7);   // col 20 of tile1 (ones col)

    float sP = 0.0f, sN = 0.0f, sQ = 0.0f, sS = 0.0f;
    for (int ks = 0; ks < KS; ++ks) {
        const float* pb = partial + (size_t)((ks * BB + b) * NT + nt) * 1536;
        sP += pb[(0 + tile) * 256 + elem];
        sN += pb[(2 + tile) * 256 + elem];
        sQ += pb[(4 + tile) * 256 + elem];
        sS += pb[5 * 256 + elemS];
    }
    const float cost_mask = (sP + sN) * (1.0f / (float)HW);
    const float num  = 2.0f * sQ;
    const float den  = sS + stgt[b * MM + m];
    const float dice = 1.0f - (num + 1.0f) / (den + 1.0f);
    out[t] = cost_mask + dice;
}

// ---------------------------------------------------------------------------
extern "C" void kernel_launch(void* const* d_in, const int* in_sizes, int n_in,
                              void* d_out, int out_size, void* d_ws, size_t ws_size,
                              hipStream_t stream) {
    (void)in_sizes; (void)n_in; (void)out_size; (void)ws_size;
    const float* x   = (const float*)d_in[0];   // outputs [4,100,256,256] f32 logits
    const float* tgt = (const float*)d_in[1];   // targets [4,20,256,256]  f32 binary
    float* out = (float*)d_out;                 // [4,100,20] f32

    float* partial = (float*)d_ws;                          // KS*BB*NT*1536 floats (~2.6 MB)
    float* stgt    = partial + (size_t)KS * BB * NT * 1536; // BB*MM floats

    cost_tgtsum  <<<BB * MM,        256, 0, stream>>>(tgt, stgt);
    cost_main    <<<BB * NT * KS,   256, 0, stream>>>(x, tgt, partial);
    cost_finalize<<<(BB * NN * MM + 255) / 256, 256, 0, stream>>>(partial, stgt, out);
}
